// MathyEmbedding_32667521253657
// MI455X (gfx1250) — compile-verified, matrix operand order checked by reading, not run on hardware
//
#include <hip/hip_runtime.h>
#include <hip/hip_bf16.h>
#include <stdint.h>

// ---------------------------------------------------------------------------
// Types for CDNA5 WMMA (wave32)
// ---------------------------------------------------------------------------
typedef __attribute__((ext_vector_type(16))) __bf16          v16bf;
typedef __attribute__((ext_vector_type(16))) unsigned short  v16u;
typedef __attribute__((ext_vector_type(8)))  unsigned short  v8u;
typedef __attribute__((ext_vector_type(8)))  float           v8f;

#define T_B   256      // time axis (reference B)
#define N_S   512      // batch axis (reference S)
#define LU    512      // LSTM_UNITS == UNITS
#define NG    2048     // 4*LU gate width
#define AST   520      // padded LDS row stride (elements)

__device__ __forceinline__ unsigned short f2bf(float f) {
  unsigned int u = __builtin_bit_cast(unsigned int, f);
  unsigned int r = u + 0x7FFFu + ((u >> 16) & 1u);   // round-to-nearest-even
  return (unsigned short)(r >> 16);
}
__device__ __forceinline__ float sigm(float x) { return 1.0f / (1.0f + __expf(-x)); }
__device__ __forceinline__ float silu(float x) { return x * sigm(x); }

// branch-free tanh: prefer hardware v_tanh_f32 when the builtin exists
__device__ __forceinline__ float fast_tanh(float x) {
#if __has_builtin(__builtin_amdgcn_tanhf)
  return __builtin_amdgcn_tanhf(x);
#elif __has_builtin(__builtin_amdgcn_tanh_f32)
  return __builtin_amdgcn_tanh_f32(x);
#else
  // tanh(x) = 2*sigmoid(2x) - 1 ; saturates correctly, no divergent branches
  return 2.0f / (1.0f + __expf(-2.0f * x)) - 1.0f;
#endif
}

__device__ __forceinline__ v16bf ld_afrag(const unsigned short* ab) {
  // per-lane A fragment: halves {0..7} then {16..23} relative to base
  v8u lo = *(const v8u*)(ab);
  v8u hi = *(const v8u*)(ab + 16);
  v16u u = __builtin_shufflevector(lo, hi, 0,1,2,3,4,5,6,7,8,9,10,11,12,13,14,15);
  return __builtin_bit_cast(v16bf, u);
}

// ---------------------------------------------------------------------------
// Pack [Wk[l]; Wr[l]]  (K=1024 x N=2048, f32) -> bf16 WMMA B-fragment layout:
//   Wp[l][nt][kt][lane][e0..15], 32B contiguous per lane.
// ---------------------------------------------------------------------------
__global__ void prepack_lstm_w(const float* __restrict__ Wk,
                               const float* __restrict__ Wr,
                               unsigned short* __restrict__ Wp) {
  int gid  = blockIdx.x * 256 + threadIdx.x;   // 3*128*32*32 threads
  int lane = gid & 31;
  int kt   = (gid >> 5) & 31;
  int nt   = (gid >> 10) & 127;
  int l    = gid >> 17;
  int n    = nt * 16 + (lane & 15);
  int kb   = kt * 32 + ((lane & 16) ? 8 : 0);
  v16u ov;
#pragma unroll
  for (int e = 0; e < 16; ++e) {
    int k = kb + ((e < 8) ? e : (8 + e));      // {0..7} then {16..23}
    float v = (k < LU) ? Wk[((size_t)l * LU + k) * NG + n]
                       : Wr[((size_t)l * LU + (k - LU)) * NG + n];
    ov[e] = f2bf(v);
  }
  *(v16u*)(Wp + ((((size_t)l * 128 + nt) * 32 + kt) * 32 + lane) * 16) = ov;
}

// Pack W_out[0:512][0:512] (f32) into same B-fragment layout (KT = 16)
__global__ void prepack_wo(const float* __restrict__ W_out,
                           unsigned short* __restrict__ Wo) {
  int gid  = blockIdx.x * 256 + threadIdx.x;   // 32*16*32 threads
  int lane = gid & 31;
  int kt   = (gid >> 5) & 15;
  int nt   = gid >> 9;
  int n    = nt * 16 + (lane & 15);
  int kb   = kt * 32 + ((lane & 16) ? 8 : 0);
  v16u ov;
#pragma unroll
  for (int e = 0; e < 16; ++e) {
    int k = kb + ((e < 8) ? e : (8 + e));
    ov[e] = f2bf(W_out[(size_t)k * LU + n]);
  }
  *(v16u*)(Wo + (((size_t)nt * 16 + kt) * 32 + lane) * 16) = ov;
}

// ---------------------------------------------------------------------------
// E_proj[j][v][u] = sum_k emb[v][k] * W_in[j*512+k][u]   (3 x 64 x 512)
// ---------------------------------------------------------------------------
__global__ void eproj_k(const float* __restrict__ emb,
                        const float* __restrict__ W_in,
                        float* __restrict__ E) {
  int gid = blockIdx.x * 256 + threadIdx.x;    // 3*64*512
  int u = gid & 511, v = (gid >> 9) & 63, j = gid >> 15;
  float s = 0.f;
  for (int k = 0; k < LU; ++k)
    s += emb[v * LU + k] * W_in[((size_t)(j * LU + k)) * LU + u];
  E[gid] = s;
}

// ---------------------------------------------------------------------------
// X0[t][n][u] = bf16( silu( E0[w0] + E1[w1] + E2[w2] + values*W_in_last + b_in ) )
// ---------------------------------------------------------------------------
__global__ void x0_k(const int* __restrict__ nodes,
                     const float* __restrict__ values,
                     const float* __restrict__ E,
                     const float* __restrict__ W_in,
                     const float* __restrict__ b_in,
                     unsigned short* __restrict__ X) {
  size_t gid = (size_t)blockIdx.x * 256 + threadIdx.x;   // 256*512*512
  int u = (int)(gid & 511);
  int n = (int)((gid >> 9) & 511);
  int t = (int)(gid >> 18);
  int base = t * N_S + n;
  int w0 = (n > 0)       ? nodes[base - 1] : 0;          // SAME pad -> token 0
  int w1 =                 nodes[base];
  int w2 = (n < N_S - 1) ? nodes[base + 1] : 0;
  float s = E[((size_t)(0 * 64 + w0)) * LU + u]
          + E[((size_t)(1 * 64 + w1)) * LU + u]
          + E[((size_t)(2 * 64 + w2)) * LU + u]
          + values[base] * W_in[(size_t)1536 * LU + u] + b_in[u];
  X[gid] = f2bf(silu(s));
}

// ---------------------------------------------------------------------------
// The whole 3-layer, 256-step LSTM + LayerNorm + q_sum in ONE kernel.
// 32 workgroups x 1024 threads (32 waves). WG b owns batch rows [16b,16b+16).
// No inter-workgroup coupling exists (rows independent; LN over features).
//  - persistent h (bf16) in LDS, c in registers (C-tile mapped)
//  - per step: z = [x_t | h] @ [Wk;Wr]  via v_wmma_f32_16x16x32_bf16
//  - wave w computes gate-aligned cols j in [16w,16w+16) of all 4 gates
// ---------------------------------------------------------------------------
__global__ __launch_bounds__(1024, 1)
void lstm_all(unsigned short* __restrict__ X,        // [256][512][512] bf16, in/out
              float* __restrict__ q_sum,             // [256][512][512] f32
              const unsigned short* __restrict__ Wp, // packed weights, 3 layers
              const float* __restrict__ bias,        // bl [3][2048]
              const float* __restrict__ h0,
              const float* __restrict__ c0,
              const float* __restrict__ lng,
              const float* __restrict__ lnb,
              float* __restrict__ h_last) {          // [512] final h row 511
  __shared__ unsigned short Ph[16 * AST];            // persistent h (bf16) 16,640B
  __shared__ float          S [16 * AST];            // x(bf16)/h(f32)      33,280B
  __shared__ float          LG[LU], LB[LU];          // LN params            4,096B
  __shared__ float          BB[NG];                  // layer bias           8,192B

  const int tid  = threadIdx.x;
  const int lane = tid & 31;
  const int w    = tid >> 5;                         // wave 0..31
  const int l15  = lane & 15;
  const int koff = (lane & 16) ? 8 : 0;
  const int hi8  = (lane & 16) ? 8 : 0;              // C/D row offset
  const int n0   = blockIdx.x * 16;
  const int j    = w * 16 + l15;                     // this wave's gate column

  // init persistent h, LN params, per-lane c registers
  for (int i = tid; i < 16 * LU; i += 1024) {
    int m = i >> 9, jj = i & 511;
    Ph[m * AST + jj] = f2bf(h0[jj]);
  }
  for (int i = tid; i < LU; i += 1024) { LG[i] = lng[i]; LB[i] = lnb[i]; }
  v8f creg;
  {
    float cv = c0[j];
#pragma unroll
    for (int r = 0; r < 8; ++r) creg[r] = cv;
  }
  __syncthreads();

  for (int l = 0; l < 3; ++l) {
    const unsigned short* Wl = Wp + (size_t)l * (128 * 32 * 32 * 16);
    for (int i = tid; i < NG; i += 1024) BB[i] = bias[l * NG + i];

    for (int t = 0; t < T_B; ++t) {
      // ---- stage x_t (16 rows x 512 bf16) into S ----
      {
        unsigned int* Sx = (unsigned int*)S;                       // 260 uints/row
        const unsigned int* Xg =
            (const unsigned int*)(X + ((size_t)t * N_S + n0) * LU);
        for (int i = tid; i < 16 * 256; i += 1024) {
          int m = i >> 8, p = i & 255;
          Sx[m * 260 + p] = Xg[m * 256 + p];
        }
      }
      __syncthreads();   // also covers BB fill (first iter) and BB reuse

      // ---- z = [x|h] @ W : 4 tiles/wave (1 per gate), K = 1024 ----
      v8f acc[4];
#pragma unroll
      for (int g = 0; g < 4; ++g) acc[g] = (v8f){0.f,0.f,0.f,0.f,0.f,0.f,0.f,0.f};

      const unsigned short* Sx16 = (const unsigned short*)S;
      for (int kt = 0; kt < 32; ++kt) {
        const unsigned short* ab = (kt < 16)
            ? (Sx16 + l15 * AST + kt * 32 + koff)
            : (Ph   + l15 * AST + (kt - 16) * 32 + koff);
        v16bf a = ld_afrag(ab);
#pragma unroll
        for (int g = 0; g < 4; ++g) {
          int nt = g * 32 + w;
          v16u bu = *(const v16u*)(Wl + (((size_t)nt * 32 + kt) * 32 + lane) * 16);
          v16bf b = __builtin_bit_cast(v16bf, bu);
          acc[g] = __builtin_amdgcn_wmma_f32_16x16x32_bf16(
              false, a, false, b, (short)0, acc[g], false, false);
        }
      }
      __syncthreads();   // all waves done reading S(x) and Ph(h_{t-1})

      // ---- gates / cell / hidden update (all in registers) ----
      {
        float bi0 = BB[j], bf0 = BB[512 + j], bg0 = BB[1024 + j], bo0 = BB[1536 + j];
#pragma unroll
        for (int r = 0; r < 8; ++r) {
          int m = r + hi8;
          float zi = acc[0][r] + bi0;
          float zf = acc[1][r] + bf0;
          float zg = acc[2][r] + bg0;
          float zo = acc[3][r] + bo0;
          float cn = sigm(zf) * creg[r] + sigm(zi) * fast_tanh(zg);
          float hn = sigm(zo) * fast_tanh(cn);
          creg[r] = cn;
          S [m * AST + j] = hn;          // f32 for LN
          Ph[m * AST + j] = f2bf(hn);    // bf16 for next step's GEMM
        }
      }
      __syncthreads();

      // ---- LayerNorm: wave w (<16) normalizes row m = w over 512 features ----
      if (w < 16) {
        int m = w;
        float hv[16], s1 = 0.f, s2 = 0.f;
#pragma unroll
        for (int e = 0; e < 16; ++e) {
          float v = S[m * AST + lane + 32 * e];
          hv[e] = v; s1 += v; s2 += v * v;
        }
#pragma unroll
        for (int off = 16; off > 0; off >>= 1) {
          s1 += __shfl_xor(s1, off, 32);
          s2 += __shfl_xor(s2, off, 32);
        }
        float mean = s1 * (1.0f / 512.0f);
        float var  = s2 * (1.0f / 512.0f) - mean * mean;
        float rstd = rsqrtf(var + 1e-3f);
        size_t rowo = ((size_t)t * N_S + n0 + m) * LU;
#pragma unroll
        for (int e = 0; e < 16; ++e) {
          int jj = lane + 32 * e;
          float y = LG[jj] * (hv[e] - mean) * rstd + LB[jj];
          X[rowo + jj] = f2bf(y);                      // input to next layer
          if (l == 0) q_sum[rowo + jj] = y;
          else        q_sum[rowo + jj] += y;
        }
      }
      __syncthreads();   // before next step's staging overwrites S
    }
  }

  // final h of last layer, batch row 511 (owned by workgroup 31, LDS row 15)
  if (blockIdx.x == 31 && tid < 512) h_last[tid] = S[15 * AST + tid];
}

// ---------------------------------------------------------------------------
// ctx_bias[u] = b_out[u] + h_last . W_out[512:1024][u] + hist . W_out[1024:1536][u]
// ---------------------------------------------------------------------------
__global__ void ctx_k(const float* __restrict__ h_last,
                      const float* __restrict__ hist,
                      const float* __restrict__ W_out,
                      const float* __restrict__ b_out,
                      float* __restrict__ ctxb) {
  int u = blockIdx.x * 256 + threadIdx.x;
  if (u >= LU) return;
  float s = b_out[u];
  for (int k = 0; k < LU; ++k) s += h_last[k] * W_out[((size_t)(512 + k)) * LU + u];
  for (int k = 0; k < LU; ++k) s += hist[k]   * W_out[((size_t)(1024 + k)) * LU + u];
  ctxb[u] = s;
}

// ---------------------------------------------------------------------------
// out[row][u] = silu( q_sum[row] @ W_out[:512] + ctx_bias[u] )
// WMMA bf16; block = 32 rows x 512 cols, 8 waves, K = 512.
// ---------------------------------------------------------------------------
__global__ __launch_bounds__(256, 1)
void out_gemm(const float* __restrict__ q_sum,
              const unsigned short* __restrict__ Wo,
              const float* __restrict__ ctxb,
              float* __restrict__ out) {
  __shared__ unsigned short A[32 * AST];
  const int tid  = threadIdx.x;
  const int lane = tid & 31;
  const int w    = tid >> 5;
  const int l15  = lane & 15;
  const int koff = (lane & 16) ? 8 : 0;
  const int hi8  = (lane & 16) ? 8 : 0;
  const size_t r0 = (size_t)blockIdx.x * 32;

  for (int i = tid; i < 32 * LU; i += 256) {
    int m = i >> 9, j = i & 511;
    A[m * AST + j] = f2bf(q_sum[(r0 + m) * LU + j]);
  }
  __syncthreads();

  const int m0 = (w & 1) * 16;
  const int c0 = (w >> 1) * 128;
  v8f acc[8];
#pragma unroll
  for (int g = 0; g < 8; ++g) acc[g] = (v8f){0.f,0.f,0.f,0.f,0.f,0.f,0.f,0.f};

  for (int kt = 0; kt < 16; ++kt) {
    v16bf a = ld_afrag(A + (m0 + l15) * AST + kt * 32 + koff);
#pragma unroll
    for (int ti = 0; ti < 8; ++ti) {
      int nt = (c0 >> 4) + ti;
      v16u bu = *(const v16u*)(Wo + (((size_t)nt * 16 + kt) * 32 + lane) * 16);
      v16bf b = __builtin_bit_cast(v16bf, bu);
      acc[ti] = __builtin_amdgcn_wmma_f32_16x16x32_bf16(
          false, a, false, b, (short)0, acc[ti], false, false);
    }
  }

#pragma unroll
  for (int ti = 0; ti < 8; ++ti) {
    int col = c0 + ti * 16 + l15;
    float cb = ctxb[col];
#pragma unroll
    for (int r = 0; r < 8; ++r) {
      int m = m0 + r + hi8;
      out[(r0 + m) * LU + col] = silu(acc[ti][r] + cb);
    }
  }
}

__global__ void tail_k(float* out, int start, int count) {
  int i = blockIdx.x * 256 + threadIdx.x;
  if (i < count) out[start + i] = 512.0f;     // second tuple output: Sn
}

// ---------------------------------------------------------------------------
extern "C" void kernel_launch(void* const* d_in, const int* in_sizes, int n_in,
                              void* d_out, int out_size, void* d_ws, size_t ws_size,
                              hipStream_t stream) {
  (void)in_sizes; (void)n_in; (void)ws_size;
  const int*   nodes  = (const int*)  d_in[0];
  const float* values = (const float*)d_in[1];
  const float* rs_h   = (const float*)d_in[2];
  const float* rs_c   = (const float*)d_in[3];
  const float* hist   = (const float*)d_in[4];
  const float* emb    = (const float*)d_in[5];
  const float* W_in   = (const float*)d_in[6];
  const float* b_in   = (const float*)d_in[7];
  const float* Wk     = (const float*)d_in[8];
  const float* Wr     = (const float*)d_in[9];
  const float* bl     = (const float*)d_in[10];
  const float* ln_g   = (const float*)d_in[11];
  const float* ln_b   = (const float*)d_in[12];
  const float* W_out  = (const float*)d_in[13];
  const float* b_out  = (const float*)d_in[14];
  float* out = (float*)d_out;

  uint8_t* ws = (uint8_t*)d_ws;
  size_t off = 0;
  auto carve = [&](size_t bytes) -> void* {
    void* p = ws + off;
    off += (bytes + 255) & ~(size_t)255;
    return p;
  };
  unsigned short* Wp     = (unsigned short*)carve((size_t)3 * 128 * 32 * 32 * 16 * 2); // 12 MB
  unsigned short* Wo     = (unsigned short*)carve((size_t)32 * 16 * 32 * 16 * 2);      // 0.5 MB
  float*          E      = (float*)         carve((size_t)3 * 64 * 512 * 4);
  float*          ctxb   = (float*)         carve(512 * 4);
  float*          h_last = (float*)         carve(512 * 4);
  unsigned short* X      = (unsigned short*)carve((size_t)T_B * N_S * LU * 2);         // 128 MB
  float*          q_sum  = (float*)         carve((size_t)T_B * N_S * LU * 4);         // 256 MB

  prepack_lstm_w<<<(3 * 128 * 32 * 32) / 256, 256, 0, stream>>>(Wk, Wr, Wp);
  prepack_wo    <<<(32 * 16 * 32) / 256,      256, 0, stream>>>(W_out, Wo);
  eproj_k       <<<(3 * 64 * 512) / 256,      256, 0, stream>>>(emb, W_in, E);
  x0_k          <<<(T_B * N_S * LU) / 256,    256, 0, stream>>>(nodes, values, E, W_in, b_in, X);
  lstm_all      <<<32, 1024, 0, stream>>>(X, q_sum, Wp, bl, rs_h, rs_c, ln_g, ln_b, h_last);
  ctx_k         <<<2, 256, 0, stream>>>(h_last, hist, W_out, b_out, ctxb);
  out_gemm      <<<(T_B * N_S) / 32, 256, 0, stream>>>(q_sum, Wo, ctxb, out);

  const int main_n = T_B * N_S * LU;
  if (out_size > main_n) {
    int tail = out_size - main_n;
    tail_k<<<(tail + 255) / 256, 256, 0, stream>>>(out, main_n, tail);
  }
}
